// MultiHeadAttention_2972117369058
// MI455X (gfx1250) — matrix operand order repaired and checked by reading, not compiled, not run on hardware
//
#include <hip/hip_runtime.h>
#include <hip/hip_bf16.h>
#include <stdint.h>

// MHA sizes (fixed by the reference)
#define HH 8
#define DD 512
#define BB 2
#define SS 2048

typedef __attribute__((ext_vector_type(16))) __bf16 v16bf;
typedef __attribute__((ext_vector_type(8)))  float  v8f;
typedef int i32x4_t __attribute__((vector_size(16)));  // matches builtin prototype

// ---- CDNA5 async global->LDS path (guarded; fallback = load + ds_store) ----
#if defined(__has_builtin)
#if __has_builtin(__builtin_amdgcn_global_load_async_to_lds_b128) && \
    __has_builtin(__builtin_amdgcn_s_wait_asynccnt)
#define ASYNC_OK 1
#endif
#endif
#ifndef ASYNC_OK
#define ASYNC_OK 0
#endif

struct Pack32 { uint4 a, b; };

// Load a 16-element bf16 fragment from two 16-byte chunks.
__device__ __forceinline__ v16bf ld_frag(const unsigned short* p0, const unsigned short* p1) {
  Pack32 t;
  t.a = *reinterpret_cast<const uint4*>(p0);
  t.b = *reinterpret_cast<const uint4*>(p1);
  return __builtin_bit_cast(v16bf, t);
}

// f32 -> bf16 round-to-nearest-even
__device__ __forceinline__ unsigned short f2bf(float f) {
  unsigned int u = __float_as_uint(f);
  u += 0x7FFFu + ((u >> 16) & 1u);
  return (unsigned short)(u >> 16);
}

// ---------------- conversion kernels ----------------

__global__ void cvt_bf16_kernel(const float* __restrict__ in,
                                unsigned short* __restrict__ out, int n) {
  int i = blockIdx.x * blockDim.x + threadIdx.x;
  if (i < n) out[i] = f2bf(in[i]);
}

// in: [z][R][C] f32  ->  out: [z][C][R] bf16   (transpose per batch z)
__global__ void cvt_t_bf16_kernel(const float* __restrict__ in,
                                  unsigned short* __restrict__ out, int R, int C) {
  long base = (long)blockIdx.z * R * C;
  int i = blockIdx.x * blockDim.x + threadIdx.x;
  if (i >= R * C) return;
  int c = i / R;
  int r = i - c * R;
  out[base + i] = f2bf(in[base + (long)r * C + c]);
}

// ---------------- LDS staging helpers ----------------
// B tile in LDS: 128 rows (N) x 64 elems (K), row stride 72 elems (pad spreads banks).
#define LDS_STRIDE 72

__device__ __forceinline__ void stage_b(const unsigned short* __restrict__ gbase,
                                        int ldb, int k0,
                                        unsigned short* __restrict__ lbuf, int tid) {
  // 128 rows * 8 chunks(16B) = 1024 chunks; 4 per thread.
#pragma unroll
  for (int i = 0; i < 4; ++i) {
    const int c  = tid + i * 256;
    const int n  = c >> 3;
    const int kc = c & 7;
    const unsigned short* gp = gbase + (long)n * ldb + k0 + kc * 8;
    unsigned short* lp = lbuf + n * LDS_STRIDE + kc * 8;
#if ASYNC_OK
    __builtin_amdgcn_global_load_async_to_lds_b128(
        (__attribute__((address_space(1))) i32x4_t*)gp,
        (__attribute__((address_space(3))) i32x4_t*)lp, 0, 0);
#else
    *reinterpret_cast<uint4*>(lp) = *reinterpret_cast<const uint4*>(gp);
#endif
  }
}

__device__ __forceinline__ void wait_stage() {
#if ASYNC_OK
  __builtin_amdgcn_s_wait_asynccnt(0);
#endif
  __syncthreads();
}

// ---------------- generic bf16 WMMA GEMM ----------------
// C[M,N] = A[M,K] * Bt[N,K]^T + bias[N]
// Block: 256 threads = 8 waves; block tile 128(M) x 128(N); wave tile 32 x 64.
// B tile double-buffered in LDS via async copies; A fragments direct from global.
// MODE 0: bf16 store at row*ldc + col
// MODE 1: bf16 store at col*ldc + row   (transposed output)
// MODE 2: f32  store at row*ldc + col
template <int MODE>
__global__ void __launch_bounds__(256)
gemm_bf16_kernel(const unsigned short* __restrict__ A, long a_bs, long a_hs, int lda,
                 const unsigned short* __restrict__ Bt, long b_bs, long b_hs, int ldb,
                 const float* __restrict__ bias, int bias_hs,
                 void* __restrict__ Cv, long c_bs, long c_hs, int ldc,
                 int K) {
  __shared__ unsigned short lb[2][128 * LDS_STRIDE];

  const int tid  = threadIdx.x;
  const int z    = blockIdx.z;
  const int b    = z / HH;
  const int h    = z - b * HH;
  const int lane = tid & 31;
  const int wv   = tid >> 5;
  const int hf   = lane >> 4;   // half-wave select
  const int l15  = lane & 15;
  const int wm   = wv & 3;      // wave M quadrant (0..3)
  const int wn   = wv >> 2;     // wave N half (0..1)
  const int m0   = blockIdx.y * 128 + wm * 32;
  const int nblk = blockIdx.x * 128;
  const int nw   = wn * 64;     // wave's column offset inside block tile

  const unsigned short* Ab = A + b * a_bs + h * a_hs;
  const unsigned short* Bb = Bt + b * b_bs + h * b_hs + (long)nblk * ldb;
  const long coff = b * c_bs + h * c_hs;

  v8f acc[2][4] = {};

  // A fragment bases: lane l15 = row, halves split K into (0..7,16..23)/(8..15,24..31)
  const unsigned short* ar0 = Ab + (long)(m0 + l15) * lda + hf * 8;
  const unsigned short* ar1 = Ab + (long)(m0 + 16 + l15) * lda + hf * 8;

  stage_b(Bb, ldb, 0, lb[0], tid);

  const int NT = K / 64;
  for (int kt = 0; kt < NT; ++kt) {
    const int cur = kt & 1;
    wait_stage();  // async copies of lb[cur] complete + all waves past last tile
    if (kt + 1 < NT) stage_b(Bb, ldb, (kt + 1) * 64, lb[cur ^ 1], tid);

    const unsigned short* lbc = lb[cur];
#pragma unroll
    for (int ks = 0; ks < 2; ++ks) {
      const int kg = kt * 64 + ks * 32;
      v16bf a0 = ld_frag(ar0 + kg, ar0 + kg + 16);
      v16bf a1 = ld_frag(ar1 + kg, ar1 + kg + 16);
#pragma unroll
      for (int j = 0; j < 4; ++j) {
        const unsigned short* lp =
            lbc + (nw + j * 16 + l15) * LDS_STRIDE + ks * 32 + hf * 8;
        v16bf wf = ld_frag(lp, lp + 16);
        acc[0][j] = __builtin_amdgcn_wmma_f32_16x16x32_bf16(
            false, a0, false, wf, (short)0, acc[0][j], false, false);
        acc[1][j] = __builtin_amdgcn_wmma_f32_16x16x32_bf16(
            false, a1, false, wf, (short)0, acc[1][j], false, false);
      }
    }
  }

#pragma unroll
  for (int im = 0; im < 2; ++im) {
#pragma unroll
    for (int j = 0; j < 4; ++j) {
      const int col = nblk + nw + j * 16 + l15;
      const float bias_v = bias ? bias[bias_hs * h + col] : 0.0f;
#pragma unroll
      for (int r = 0; r < 8; ++r) {
        const int row = m0 + im * 16 + r + hf * 8;  // C layout: vgpr r + 8*half
        const float val = acc[im][j][r] + bias_v;
        if (MODE == 0)
          ((unsigned short*)Cv)[coff + (long)row * ldc + col] = f2bf(val);
        else if (MODE == 1)
          ((unsigned short*)Cv)[coff + (long)col * ldc + row] = f2bf(val);
        else
          ((float*)Cv)[coff + (long)row * ldc + col] = val;
      }
    }
  }
}

// ---------------- attention: QK^T + mask + softmax -> P (bf16) ----------------

__device__ __forceinline__ void qk_tile_acc(const unsigned short* __restrict__ Aq,
                                            const unsigned short* __restrict__ Bk,
                                            int m0, int t0, int l15, int hf,
                                            v8f acc[4]) {
  const unsigned short* arowp = Aq + (long)(m0 + l15) * DD + hf * 8;
#pragma unroll 4
  for (int k0 = 0; k0 < DD; k0 += 32) {
    v16bf af = ld_frag(arowp + k0, arowp + k0 + 16);
#pragma unroll
    for (int j = 0; j < 4; ++j) {
      const unsigned short* bp = Bk + (long)(t0 + j * 16 + l15) * DD + k0 + hf * 8;
      v16bf kf = ld_frag(bp, bp + 16);
      acc[j] = __builtin_amdgcn_wmma_f32_16x16x32_bf16(
          false, af, false, kf, (short)0, acc[j], false, false);
    }
  }
}

// Grid: (SS/128, B*H). 8 waves/block, each wave owns a 16-row query strip.
__global__ void __launch_bounds__(256)
attn_softmax_kernel(const unsigned short* __restrict__ Qp,   // [B,H,S,D]
                    const unsigned short* __restrict__ Kp,   // [B,H,S,D]
                    const int* __restrict__ mask,            // [S,S]
                    unsigned short* __restrict__ P) {        // [B,H,S,S]
  const int z    = blockIdx.y;
  const int lane = threadIdx.x & 31;
  const int wv   = threadIdx.x >> 5;
  const int hf   = lane >> 4;
  const int l15  = lane & 15;
  const int m0   = blockIdx.x * 128 + wv * 16;

  const unsigned short* Aq = Qp + (long)z * SS * DD;
  const unsigned short* Bk = Kp + (long)z * SS * DD;
  unsigned short* Pp = P + (long)z * SS * SS;

  const float scale = 0.044194173824159216f;  // 1/sqrt(512)

  float mrow[8], lrow[8];
#pragma unroll
  for (int r = 0; r < 8; ++r) { mrow[r] = -3.0e38f; lrow[r] = 0.0f; }

  // Pass 1: streaming row max / sum-exp over all keys (online softmax stats).
  for (int t0 = 0; t0 < SS; t0 += 64) {
    v8f acc[4] = {};
    qk_tile_acc(Aq, Bk, m0, t0, l15, hf, acc);
    float s[4][8];
#pragma unroll
    for (int j = 0; j < 4; ++j) {
      const int col = t0 + j * 16 + l15;
#pragma unroll
      for (int r = 0; r < 8; ++r) {
        const int row = m0 + r + hf * 8;
        float vv = acc[j][r] * scale;
        if (mask[(long)row * SS + col] == 1) vv = -1.0e9f;
        s[j][r] = vv;
      }
    }
#pragma unroll
    for (int r = 0; r < 8; ++r) {
      float tm = fmaxf(fmaxf(s[0][r], s[1][r]), fmaxf(s[2][r], s[3][r]));
      // xor masks 1,2,4,8 stay inside each 16-lane half (rows live per half)
#pragma unroll
      for (int off = 1; off < 16; off <<= 1) tm = fmaxf(tm, __shfl_xor(tm, off, 32));
      const float mnew = fmaxf(mrow[r], tm);
      float sum = 0.0f;
#pragma unroll
      for (int j = 0; j < 4; ++j) sum += __expf(s[j][r] - mnew);
#pragma unroll
      for (int off = 1; off < 16; off <<= 1) sum += __shfl_xor(sum, off, 32);
      lrow[r] = lrow[r] * __expf(mrow[r] - mnew) + sum;
      mrow[r] = mnew;
    }
  }

  float invl[8];
#pragma unroll
  for (int r = 0; r < 8; ++r) invl[r] = 1.0f / lrow[r];

  // Pass 2: recompute scores, normalize, emit P in bf16.
  for (int t0 = 0; t0 < SS; t0 += 64) {
    v8f acc[4] = {};
    qk_tile_acc(Aq, Bk, m0, t0, l15, hf, acc);
#pragma unroll
    for (int j = 0; j < 4; ++j) {
      const int col = t0 + j * 16 + l15;
#pragma unroll
      for (int r = 0; r < 8; ++r) {
        const int row = m0 + r + hf * 8;
        float vv = acc[j][r] * scale;
        if (mask[(long)row * SS + col] == 1) vv = -1.0e9f;
        const float p = __expf(vv - mrow[r]) * invl[r];
        Pp[(long)row * SS + col] = f2bf(p);
      }
    }
  }
}

// ---------------- launcher ----------------

extern "C" void kernel_launch(void* const* d_in, const int* in_sizes, int n_in,
                              void* d_out, int out_size, void* d_ws, size_t ws_size,
                              hipStream_t stream) {
  (void)in_sizes; (void)n_in; (void)out_size; (void)ws_size;
  const float* q    = (const float*)d_in[0];
  const float* k    = (const float*)d_in[1];
  const float* v    = (const float*)d_in[2];
  const int*   mask = (const int*)d_in[3];
  const float* Wq   = (const float*)d_in[4];
  const float* bq   = (const float*)d_in[5];
  const float* Wk   = (const float*)d_in[6];
  const float* bk   = (const float*)d_in[7];
  const float* Wv   = (const float*)d_in[8];
  const float* bv   = (const float*)d_in[9];
  const float* Wo   = (const float*)d_in[10];
  const float* bo   = (const float*)d_in[11];
  float* out = (float*)d_out;

  char* ws = (char*)d_ws;
  size_t off = 0;
  auto carve = [&](size_t elems) -> unsigned short* {
    unsigned short* p = (unsigned short*)(ws + off);
    off += (elems * sizeof(unsigned short) + 255) & ~(size_t)255;
    return p;
  };

  const size_t nQKV  = (size_t)BB * SS * DD;       // 2.10M
  const size_t nWh   = (size_t)HH * DD * DD;       // 2.10M
  const size_t nWo   = (size_t)DD * HH * DD;       // 2.10M
  const size_t nProj = (size_t)BB * HH * SS * DD;  // 16.8M
  const size_t nP    = (size_t)BB * HH * SS * SS;  // 67.1M

  unsigned short* qb  = carve(nQKV);
  unsigned short* kb  = carve(nQKV);
  unsigned short* vb  = carve(nQKV);
  unsigned short* Wqt = carve(nWh);
  unsigned short* Wkt = carve(nWh);
  unsigned short* Wvt = carve(nWh);
  unsigned short* Wot = carve(nWo);
  unsigned short* Qp  = carve(nProj);   // [B,H,S,D] row-major
  unsigned short* Kp  = carve(nProj);   // [B,H,S,D] row-major
  unsigned short* Vpt = carve(nProj);   // [B,H,D,S] (transposed for P*V B-operand)
  unsigned short* P   = carve(nP);      // [B,H,S,S] softmax probs
  unsigned short* Oc  = carve(nProj);   // [B,S,H*D] concat layout

  // 1) activations f32 -> bf16
  {
    int n = (int)nQKV;
    dim3 g((n + 255) / 256);
    cvt_bf16_kernel<<<g, 256, 0, stream>>>(q, qb, n);
    cvt_bf16_kernel<<<g, 256, 0, stream>>>(k, kb, n);
    cvt_bf16_kernel<<<g, 256, 0, stream>>>(v, vb, n);
  }
  // 2) weights: transpose to [N][K] + convert to bf16
  {
    dim3 g((DD * DD + 255) / 256, 1, HH);
    cvt_t_bf16_kernel<<<g, 256, 0, stream>>>(Wq, Wqt, DD, DD);
    cvt_t_bf16_kernel<<<g, 256, 0, stream>>>(Wk, Wkt, DD, DD);
    cvt_t_bf16_kernel<<<g, 256, 0, stream>>>(Wv, Wvt, DD, DD);
    dim3 go((HH * DD * DD + 255) / 256, 1, 1);
    cvt_t_bf16_kernel<<<go, 256, 0, stream>>>(Wo, Wot, HH * DD, DD);
  }
  // 3) per-head projections (Q,K row-major; V transposed)
  {
    dim3 g(DD / 128, SS / 128, BB * HH);
    gemm_bf16_kernel<0><<<g, 256, 0, stream>>>(
        qb, (long)SS * DD, 0, DD, Wqt, 0, (long)DD * DD, DD, bq, DD,
        Qp, (long)HH * SS * DD, (long)SS * DD, DD, DD);
    gemm_bf16_kernel<0><<<g, 256, 0, stream>>>(
        kb, (long)SS * DD, 0, DD, Wkt, 0, (long)DD * DD, DD, bk, DD,
        Kp, (long)HH * SS * DD, (long)SS * DD, DD, DD);
    gemm_bf16_kernel<1><<<g, 256, 0, stream>>>(
        vb, (long)SS * DD, 0, DD, Wvt, 0, (long)DD * DD, DD, bv, DD,
        Vpt, (long)HH * DD * SS, (long)DD * SS, SS, DD);
  }
  // 4) scores + mask + online softmax -> P
  {
    dim3 g(SS / 128, BB * HH);
    attn_softmax_kernel<<<g, 256, 0, stream>>>(Qp, Kp, mask, P);
  }
  // 5) P @ V -> Oc in concat layout [B,S,H*D]
  {
    dim3 g(DD / 128, SS / 128, BB * HH);
    gemm_bf16_kernel<0><<<g, 256, 0, stream>>>(
        P, (long)HH * SS * SS, (long)SS * SS, SS,
        Vpt, (long)HH * DD * SS, (long)DD * SS, SS, nullptr, 0,
        Oc, (long)SS * HH * DD, DD, HH * DD, SS);
  }
  // 6) Oc @ Wo + bo -> out (f32)
  {
    dim3 g(DD / 128, (BB * SS) / 128, 1);
    gemm_bf16_kernel<2><<<g, 256, 0, stream>>>(
        Oc, 0, 0, HH * DD, Wot, 0, 0, HH * DD, bo, 0,
        out, 0, 0, DD, HH * DD);
  }
}
